// RAttention_22033182228882
// MI455X (gfx1250) — compile-verified, matrix operand order checked by reading, not run on hardware
//
#include <hip/hip_runtime.h>
#include <hip/hip_bf16.h>

typedef __attribute__((ext_vector_type(8)))  __bf16 v8bf;
typedef __attribute__((ext_vector_type(16))) __bf16 v16bf;
typedef __attribute__((ext_vector_type(8)))  float  v8f;

#define L_SEQ 2048
#define CDIM  1024
#define HD    64
#define NH    16
#define BATCH 2

// ---------------------------------------------------------------------------
// WMMA wrapper: D = A(16x32 bf16) * B(32x16 bf16) + C(16x16 f32)
// ---------------------------------------------------------------------------
__device__ __forceinline__ v8f wmma_bf16(v16bf a, v16bf b, v8f c) {
  return __builtin_amdgcn_wmma_f32_16x16x32_bf16(
      /*neg_a=*/false, a, /*neg_b=*/false, b,
      /*c_mod=*/(short)0, c, /*reuse_a=*/false, /*reuse_b=*/false);
}

// ---------------------------------------------------------------------------
// CDNA5 async global->LDS copy (ASYNCcnt-tracked, ISA 15.18.3 op 98).
// VDST is the per-lane LDS byte address, VADDR the 64-bit global address.
// Toolchain-portable via inline asm (builtin arity differs across toolchains).
// ---------------------------------------------------------------------------
__device__ __forceinline__ unsigned lds_addr(const void* p) {
  // flat LDS aperture: low 32 bits of the generic address are the LDS offset
  return (unsigned)(unsigned long long)p;
}
__device__ __forceinline__ void async_copy_b128(unsigned lds_off,
                                                const void* gptr) {
  asm volatile("global_load_async_to_lds_b128 %0, %1, off"
               :: "v"(lds_off), "v"((unsigned long long)gptr)
               : "memory");
}
__device__ __forceinline__ void wait_async0() {
  asm volatile("s_wait_asynccnt 0x0" ::: "memory");
}

// ---------------------------------------------------------------------------
// Fragment loader for 16-bit A/B operands per CDNA5 ISA 7.12.2:
//   lanes 0-15 : row/col = lane,      elements 0-7 -> K 0..7,  8-15 -> K 16..23
//   lanes 16-31: row/col = lane-16,   elements 0-7 -> K 8..15, 8-15 -> K 24..31
// `base` is row-major [row][k] with leading dim `ld` (multiple of 8).
// ---------------------------------------------------------------------------
__device__ __forceinline__ v16bf load_frag(const __bf16* base, int ld, int row,
                                           int k0, int ln, int half) {
  const __bf16* p = base + (size_t)(row + ln) * ld + k0 + half * 8;
  v8bf lo = *(const v8bf*)p;         // K = k0 + half*8 + [0..7]
  v8bf hi = *(const v8bf*)(p + 16);  // K = k0 + 16 + half*8 + [0..7]
  v16bf r;
#pragma unroll
  for (int i = 0; i < 8; ++i) { r[i] = lo[i]; r[i + 8] = hi[i]; }
  return r;
}

// ---------------------------------------------------------------------------
// Kernel 1: fp32 -> bf16 weight conversion
// ---------------------------------------------------------------------------
__global__ void f32_to_bf16_kernel(const float* __restrict__ src,
                                   __bf16* __restrict__ dst, int n) {
  int i = blockIdx.x * blockDim.x + threadIdx.x;
  if (i < n) dst[i] = (__bf16)src[i];
}

// ---------------------------------------------------------------------------
// Kernel 2: fused QKV projection.  x is (L,B,C) fp32 -> rows r=(l*B+b) of a
// 4096x1024 matrix.  Block tile 64(M) x 128(N), 8 waves, each wave 16x64.
// Output Q/K/V bf16 in (B,H,L,hd).
// ---------------------------------------------------------------------------
__global__ __launch_bounds__(256) void qkv_gemm_kernel(
    const float* __restrict__ x,
    const __bf16* __restrict__ Wq, const __bf16* __restrict__ Wk,
    const __bf16* __restrict__ Wv,
    const float* __restrict__ bq, const float* __restrict__ bk,
    const float* __restrict__ bv,
    __bf16* __restrict__ Qo, __bf16* __restrict__ Ko, __bf16* __restrict__ Vo) {
  __shared__ __bf16 As[64 * 32];     // [m][k]
  __shared__ __bf16 Bts[128 * 32];   // transposed: [n][k]

  const int tid  = threadIdx.x;
  const int lane = tid & 31;
  const int wave = tid >> 5;
  const int half = lane >> 4;
  const int ln   = lane & 15;

  const int blockM = blockIdx.y * 64;
  const int blockN = blockIdx.x * 128;
  const int which  = blockIdx.z;
  const __bf16* W   = (which == 0) ? Wq : (which == 1) ? Wk : Wv;
  const float* bias = (which == 0) ? bq : (which == 1) ? bk : bv;
  __bf16* Out       = (which == 0) ? Qo : (which == 1) ? Ko : Vo;

  const int mo = (wave & 3) * 16;
  const int no = (wave >> 2) * 64;

  v8f acc0 = {}, acc1 = {}, acc2 = {}, acc3 = {};

  for (int kk = 0; kk < CDIM; kk += 32) {
    __syncthreads();
    {  // stage A: 64x32, convert fp32 -> bf16 (8 elems/thread, contiguous)
      int idx = tid * 8;
      int row = idx >> 5;
      int col = idx & 31;
      const float* src = x + (size_t)(blockM + row) * CDIM + kk + col;
#pragma unroll
      for (int i = 0; i < 8; ++i) As[row * 32 + col + i] = (__bf16)src[i];
    }
#pragma unroll
    for (int j = 0; j < 2; ++j) {  // stage B transposed: Bts[n][k]=W[kk+k][bn+n]
      int u  = tid * 2 + j;        // 0..511
      int k  = u >> 4;             // 0..31
      int n0 = (u & 15) * 8;
      v8bf wv = *(const v8bf*)(W + (size_t)(kk + k) * CDIM + blockN + n0);
#pragma unroll
      for (int i = 0; i < 8; ++i) Bts[(n0 + i) * 32 + k] = wv[i];
    }
    __syncthreads();

    v16bf a  = load_frag(As, 32, mo, 0, ln, half);
    v16bf b0 = load_frag(Bts, 32, no + 0,  0, ln, half);
    v16bf b1 = load_frag(Bts, 32, no + 16, 0, ln, half);
    v16bf b2 = load_frag(Bts, 32, no + 32, 0, ln, half);
    v16bf b3 = load_frag(Bts, 32, no + 48, 0, ln, half);
    acc0 = wmma_bf16(a, b0, acc0);
    acc1 = wmma_bf16(a, b1, acc1);
    acc2 = wmma_bf16(a, b2, acc2);
    acc3 = wmma_bf16(a, b3, acc3);
  }

  // D layout: lanes 0-15 hold rows 0-7 (VGPR v -> M=v), lanes 16-31 rows 8-15.
#pragma unroll
  for (int t = 0; t < 4; ++t) {
    v8f acc = (t == 0) ? acc0 : (t == 1) ? acc1 : (t == 2) ? acc2 : acc3;
    int nc = blockN + no + t * 16 + ln;
    float bb = bias[nc];
    int hh = nc >> 6, dd = nc & 63;
#pragma unroll
    for (int v = 0; v < 8; ++v) {
      int r = blockM + mo + v + 8 * half;
      int l = r >> 1, b_ = r & 1;   // row r = l*B + b
      Out[(((size_t)(b_ * NH + hh)) * L_SEQ + l) * HD + dd] =
          (__bf16)(acc[v] + bb);
    }
  }
}

// ---------------------------------------------------------------------------
// Kernel 3: flash attention with relative-position bias.
// grid = (L/64 q-tiles, B*H); block = 128 (4 waves, 16 q rows each).
// K chunk is staged ONCE per workgroup via async global->LDS b128 copies
// (removes the 4x redundant per-wave global reads); V chunk is transposed
// into LDS so P*V B-fragments are contiguous 16B ds_loads.
// ---------------------------------------------------------------------------
__global__ __launch_bounds__(128) void flash_attn_kernel(
    const __bf16* __restrict__ Q, const __bf16* __restrict__ K,
    const __bf16* __restrict__ V, const float* __restrict__ rel,
    __bf16* __restrict__ attnOut) {
  __shared__ __bf16 Ks[32 * HD];     // K chunk, row-major [k][d] (async-staged)
  __shared__ __bf16 Vt[HD * 32];     // V chunk transposed: [d][k]
  __shared__ __bf16 Pb[4][16 * 32];  // per-wave probability tile

  const int tid  = threadIdx.x;
  const int lane = tid & 31;
  const int wave = tid >> 5;
  const int half = lane >> 4;
  const int ln   = lane & 15;

  const int q0 = blockIdx.x * 64 + wave * 16;
  const int bh = blockIdx.y;
  const int b  = bh >> 4;
  const int h  = bh & 15;

  const __bf16* Qp = Q + (size_t)bh * L_SEQ * HD;
  const __bf16* Kp = K + (size_t)bh * L_SEQ * HD;
  const __bf16* Vp = V + (size_t)bh * L_SEQ * HD;

  // Q fragments stay resident for the whole pass (hd=64 -> two K=32 chunks)
  v16bf aQ0 = load_frag(Qp, HD, q0, 0,  ln, half);
  v16bf aQ1 = load_frag(Qp, HD, q0, 32, ln, half);

  v8f o0 = {}, o1 = {}, o2 = {}, o3 = {};
  float rm[8], rl[8];
#pragma unroll
  for (int v = 0; v < 8; ++v) { rm[v] = -1e30f; rl[v] = 0.0f; }

  for (int kc = 0; kc < L_SEQ; kc += 32) {
    __syncthreads();
    // async-stage K chunk (pure copy: 32x64 bf16, 16B per unit, ASYNCcnt)
#pragma unroll
    for (int j = 0; j < 2; ++j) {
      int u  = tid * 2 + j;        // 0..255
      int k  = u >> 3;             // 0..31
      int d0 = (u & 7) * 8;
      async_copy_b128(lds_addr(&Ks[k * HD + d0]),
                      Kp + (size_t)(kc + k) * HD + d0);
    }
#pragma unroll
    for (int j = 0; j < 2; ++j) {  // stage Vt[d][k] = V[kc+k][d] (2B scatter)
      int u  = tid * 2 + j;        // 0..255
      int k  = u >> 3;             // 0..31
      int d0 = (u & 7) * 8;
      v8bf vv = *(const v8bf*)(Vp + (size_t)(kc + k) * HD + d0);
#pragma unroll
      for (int i = 0; i < 8; ++i) Vt[(d0 + i) * 32 + k] = vv[i];
    }
    // prefetch next chunk while this one is consumed (global_prefetch_b8)
    if (kc + 32 < L_SEQ) {
      const char* nk = (const char*)(Kp + (size_t)(kc + 32) * HD);
      const char* nv = (const char*)(Vp + (size_t)(kc + 32) * HD);
      __builtin_prefetch(nk + tid * 32, 0, 1);
      __builtin_prefetch(nv + tid * 32, 0, 1);
    }
    wait_async0();      // async LDS writes landed
    __syncthreads();    // visible to all 4 waves

    // S = Q K^T : B-fragment lane n = key index, elements = d (contiguous)
    v8f s0 = {}, s1 = {};
    {
      v16bf b0 = load_frag(Ks, HD, 0,  0,  ln, half);
      v16bf b1 = load_frag(Ks, HD, 0,  32, ln, half);
      s0 = wmma_bf16(aQ0, b0, s0);
      s0 = wmma_bf16(aQ1, b1, s0);
      v16bf b2 = load_frag(Ks, HD, 16, 0,  ln, half);
      v16bf b3 = load_frag(Ks, HD, 16, 32, ln, half);
      s1 = wmma_bf16(aQ0, b2, s1);
      s1 = wmma_bf16(aQ1, b3, s1);
    }

    __bf16* Pw = &Pb[wave][0];
#pragma unroll
    for (int v = 0; v < 8; ++v) {
      int m = v + 8 * half;
      int q = q0 + m;
      // scores = (QK^T + rel[k-q+L-1]) * hd^-0.5   (scale applied after bias)
      float x0 = (s0[v] + rel[kc + ln - q + (L_SEQ - 1)]) * 0.125f;
      float x1 = (s1[v] + rel[kc + 16 + ln - q + (L_SEQ - 1)]) * 0.125f;
      // row max across the 16-lane group holding this row
      float t = fmaxf(x0, x1);
      t = fmaxf(t, __shfl_xor(t, 1, 32));
      t = fmaxf(t, __shfl_xor(t, 2, 32));
      t = fmaxf(t, __shfl_xor(t, 4, 32));
      t = fmaxf(t, __shfl_xor(t, 8, 32));
      float nm = fmaxf(rm[v], t);
      float sc = __expf(rm[v] - nm);
      rm[v] = nm;
      float p0 = __expf(x0 - nm);
      float p1 = __expf(x1 - nm);
      float rs = p0 + p1;
      rs += __shfl_xor(rs, 1, 32);
      rs += __shfl_xor(rs, 2, 32);
      rs += __shfl_xor(rs, 4, 32);
      rs += __shfl_xor(rs, 8, 32);
      rl[v] = rl[v] * sc + rs;
      o0[v] *= sc; o1[v] *= sc; o2[v] *= sc; o3[v] *= sc;
      // park P in LDS so it can re-enter as an A fragment (layout swap)
      Pw[m * 32 + ln]      = (__bf16)p0;
      Pw[m * 32 + 16 + ln] = (__bf16)p1;
    }

    // O += P(16x32) * V(32x64) : B-fragments from transposed Vt are
    // contiguous 16B ds_loads. Same-wave LDS ops are in-order (DScnt).
    v16bf aP  = load_frag(Pw, 32, 0,  0, ln, half);
    v16bf bv0 = load_frag(Vt, 32, 0,  0, ln, half);
    v16bf bv1 = load_frag(Vt, 32, 16, 0, ln, half);
    v16bf bv2 = load_frag(Vt, 32, 32, 0, ln, half);
    v16bf bv3 = load_frag(Vt, 32, 48, 0, ln, half);
    o0 = wmma_bf16(aP, bv0, o0);
    o1 = wmma_bf16(aP, bv1, o1);
    o2 = wmma_bf16(aP, bv2, o2);
    o3 = wmma_bf16(aP, bv3, o3);
  }

  // normalize and write attn (B,L,C) bf16
#pragma unroll
  for (int v = 0; v < 8; ++v) {
    int m = v + 8 * half;
    int q = q0 + m;
    float inv = 1.0f / rl[v];
    size_t rowBase = ((size_t)(b * L_SEQ + q)) * CDIM + h * HD;
    attnOut[rowBase + 0 * 16 + ln] = (__bf16)(o0[v] * inv);
    attnOut[rowBase + 1 * 16 + ln] = (__bf16)(o1[v] * inv);
    attnOut[rowBase + 2 * 16 + ln] = (__bf16)(o2[v] * inv);
    attnOut[rowBase + 3 * 16 + ln] = (__bf16)(o3[v] * inv);
  }
}

// ---------------------------------------------------------------------------
// Kernel 4: output projection attn(B,L,C)bf16 @ Wo + bo -> out (L,B,C) fp32.
// A tile is a pure bf16 copy -> async global->LDS b128.
// ---------------------------------------------------------------------------
__global__ __launch_bounds__(256) void oproj_gemm_kernel(
    const __bf16* __restrict__ attn, const __bf16* __restrict__ W,
    const float* __restrict__ bo, float* __restrict__ out) {
  __shared__ __bf16 As[64 * 32];
  __shared__ __bf16 Bts[128 * 32];

  const int tid  = threadIdx.x;
  const int lane = tid & 31;
  const int wave = tid >> 5;
  const int half = lane >> 4;
  const int ln   = lane & 15;

  const int blockM = blockIdx.y * 64;
  const int blockN = blockIdx.x * 128;
  const int mo = (wave & 3) * 16;
  const int no = (wave >> 2) * 64;

  v8f acc0 = {}, acc1 = {}, acc2 = {}, acc3 = {};

  for (int kk = 0; kk < CDIM; kk += 32) {
    __syncthreads();
    {  // async-stage A tile (pure bf16 copy, one b128 per thread)
      int idx = tid * 8;
      int row = idx >> 5;
      int col = idx & 31;
      async_copy_b128(lds_addr(&As[row * 32 + col]),
                      attn + (size_t)(blockM + row) * CDIM + kk + col);
    }
#pragma unroll
    for (int j = 0; j < 2; ++j) {  // stage B transposed (2B scatter)
      int u  = tid * 2 + j;
      int k  = u >> 4;
      int n0 = (u & 15) * 8;
      v8bf wv = *(const v8bf*)(W + (size_t)(kk + k) * CDIM + blockN + n0);
#pragma unroll
      for (int i = 0; i < 8; ++i) Bts[(n0 + i) * 32 + k] = wv[i];
    }
    wait_async0();
    __syncthreads();

    v16bf a  = load_frag(As, 32, mo, 0, ln, half);
    v16bf b0 = load_frag(Bts, 32, no + 0,  0, ln, half);
    v16bf b1 = load_frag(Bts, 32, no + 16, 0, ln, half);
    v16bf b2 = load_frag(Bts, 32, no + 32, 0, ln, half);
    v16bf b3 = load_frag(Bts, 32, no + 48, 0, ln, half);
    acc0 = wmma_bf16(a, b0, acc0);
    acc1 = wmma_bf16(a, b1, acc1);
    acc2 = wmma_bf16(a, b2, acc2);
    acc3 = wmma_bf16(a, b3, acc3);
  }

#pragma unroll
  for (int t = 0; t < 4; ++t) {
    v8f acc = (t == 0) ? acc0 : (t == 1) ? acc1 : (t == 2) ? acc2 : acc3;
    int nc = blockN + no + t * 16 + ln;
    float bb = bo[nc];
#pragma unroll
    for (int v = 0; v < 8; ++v) {
      int r  = blockM + mo + v + 8 * half;  // attn row r = b*L + l
      int b_ = r >> 11;                     // / L_SEQ
      int l  = r & (L_SEQ - 1);
      out[((size_t)(l * BATCH + b_)) * CDIM + nc] = acc[v] + bb;
    }
  }
}

// ---------------------------------------------------------------------------
// Host launcher
// ---------------------------------------------------------------------------
extern "C" void kernel_launch(void* const* d_in, const int* in_sizes, int n_in,
                              void* d_out, int out_size, void* d_ws,
                              size_t ws_size, hipStream_t stream) {
  const float* x   = (const float*)d_in[0];
  const float* Wq  = (const float*)d_in[1];
  const float* bq  = (const float*)d_in[2];
  const float* Wk  = (const float*)d_in[3];
  const float* bk  = (const float*)d_in[4];
  const float* Wv  = (const float*)d_in[5];
  const float* bv  = (const float*)d_in[6];
  const float* Wo  = (const float*)d_in[7];
  const float* bo  = (const float*)d_in[8];
  const float* rel = (const float*)d_in[9];

  char* ws = (char*)d_ws;
  const size_t MB = 1u << 20;
  __bf16* WqB = (__bf16*)(ws + 0 * MB);   // 2 MB each
  __bf16* WkB = (__bf16*)(ws + 2 * MB);
  __bf16* WvB = (__bf16*)(ws + 4 * MB);
  __bf16* WoB = (__bf16*)(ws + 6 * MB);
  __bf16* Qw  = (__bf16*)(ws + 8 * MB);   // 8 MB each, (B,H,L,hd)
  __bf16* Kw  = (__bf16*)(ws + 16 * MB);
  __bf16* Vw  = (__bf16*)(ws + 24 * MB);
  __bf16* Aw  = (__bf16*)(ws + 32 * MB);  // attn (B,L,C)

  const int NW = CDIM * CDIM;  // 1M elements per weight
  f32_to_bf16_kernel<<<NW / 256, 256, 0, stream>>>(Wq, WqB, NW);
  f32_to_bf16_kernel<<<NW / 256, 256, 0, stream>>>(Wk, WkB, NW);
  f32_to_bf16_kernel<<<NW / 256, 256, 0, stream>>>(Wv, WvB, NW);
  f32_to_bf16_kernel<<<NW / 256, 256, 0, stream>>>(Wo, WoB, NW);

  // M = B*L = 4096 rows, N = 1024 cols; block tile 64x128; z selects Q/K/V
  qkv_gemm_kernel<<<dim3(CDIM / 128, (BATCH * L_SEQ) / 64, 3), 256, 0, stream>>>(
      x, WqB, WkB, WvB, bq, bk, bv, Qw, Kw, Vw);

  flash_attn_kernel<<<dim3(L_SEQ / 64, BATCH * NH), 128, 0, stream>>>(
      Qw, Kw, Vw, rel, Aw);

  oproj_gemm_kernel<<<dim3(CDIM / 128, (BATCH * L_SEQ) / 64), 256, 0, stream>>>(
      Aw, WoB, bo, (float*)d_out);
}
